// TextDecoder_3178275799703
// MI455X (gfx1250) — matrix operand order repaired and checked by reading, not compiled, not run on hardware
//
#include <hip/hip_runtime.h>
#include <hip/hip_bf16.h>
#include <math.h>

// ---------------------------------------------------------------------------
// Autoregressive 2-layer GRU decoder for MI455X (gfx1250, wave32, WMMA).
//
//  * One-time setup: weights -> bf16; precompute
//      h0  = z @ z2h_w.T + b,   giz0 = z @ w_ih0[:,H:].T + b_ih0,
//      Etab= swish(embed) @ w_ih0[:,:H].T   (cell-0 input GEMM becomes a
//      128-row table lookup),   oz = z @ h2o_w[:,H:].T + h2o_b.
//  * Per step (64x): cell0 (1 WMMA GEMM + fused gates), cell1 (2 WMMA GEMMs +
//    fused gates), output GEMM + argmax. Hidden state double-buffered
//    (fp32 master + bf16 shadow).
//  * Weight B-tiles staged per-block into LDS with GLOBAL_LOAD_ASYNC_TO_LDS_B128
//    (ASYNCcnt-tracked, 3 buffers, 2 tiles in flight, one barrier/iter):
//    8x less L2 weight traffic, fragments become contiguous 32B ds reads.
//  * A-fragment global loads are issued before the async-wait/barrier so their
//    latency overlaps the synchronization.
// ---------------------------------------------------------------------------

#define B_  4096
#define H_  1024
#define L_  256
#define C_  128
#define T_  64
#define G_  3072   // 3*H

typedef __attribute__((ext_vector_type(16))) __bf16 v16bf;
typedef __attribute__((ext_vector_type(8)))  float  v8f;
typedef __attribute__((ext_vector_type(16))) unsigned short v16us;
typedef __attribute__((ext_vector_type(8)))  unsigned short v8us;

__device__ __forceinline__ unsigned short f2bf(float x) {
    unsigned int u = __float_as_uint(x);
    unsigned int r = u + 0x7FFFu + ((u >> 16) & 1u);   // round-to-nearest-even
    return (unsigned short)(r >> 16);
}

__device__ __forceinline__ float sigm(float x) {
    return 1.0f / (1.0f + __expf(-x));
}

// A fragment: 16x32 bf16 (MxK). lane L: row = mtile + (L&15); hi = L>>4.
// elems 0..15 : K = k0 + hi*16 + 0..15
__device__ __forceinline__ v16bf load_a_frag(const unsigned short* __restrict__ A,
                                             int row, int ld, int k0, int lane) {
    const int hi = lane >> 4;
    const unsigned short* p = A + (size_t)row * ld + k0 + hi * 16;
    union { v16bf bf; v8us h[2]; } u;
    u.h[0] = *(const v8us*)(p);
    u.h[1] = *(const v8us*)(p + 8);
    return u.bf;
}

// B fragment from global: B[k][n] = W[jrow(n)][k], W row-major [*, ld].
__device__ __forceinline__ v16bf load_b_frag(const unsigned short* __restrict__ W,
                                             int jrow, int ld, int k0, int lane) {
    const unsigned short* p = W + (size_t)jrow * ld + k0 + ((lane >> 4) << 4);
    union { v16bf bf; v16us us; } u;
    u.us = *(const v16us*)p;
    return u.bf;
}

// B fragment from an LDS tile stored in fragment order (32B slots).
__device__ __forceinline__ v16bf lds_b_frag(const unsigned short* base, int slot) {
    union { v16bf bf; v16us us; } u;
    u.us = *(const v16us*)(base + slot * 16);
    return u.bf;
}

// One 16-byte async global->LDS copy (per-lane addressing, ASYNCcnt-tracked).
__device__ __forceinline__ void async_copy16(unsigned int lds_off,
                                             const unsigned short* gsrc) {
    asm volatile("global_load_async_to_lds_b128 %0, %1, off"
                 :: "v"(lds_off), "v"(gsrc) : "memory");
}

__device__ __forceinline__ void wait_async_le1() {
    asm volatile("s_wait_asynccnt 0x1" ::: "memory");
}
__device__ __forceinline__ void wait_async_le2() {
    asm volatile("s_wait_asynccnt 0x2" ::: "memory");
}

__device__ __forceinline__ unsigned int lds_addr_of(const void* p) {
    // low 32 bits of a generic shared-space address = wave-relative LDS offset
    return (unsigned int)(unsigned long long)p;
}

#define WMMA_BF16(a, b, c) \
    __builtin_amdgcn_wmma_f32_16x16x32_bf16(false, (a), false, (b), (short)0, (c), false, false)

// ------------------------------- setup kernels -----------------------------

__global__ void k_convert_bf16(const float* __restrict__ src,
                               unsigned short* __restrict__ dst, int n) {
    int i = blockIdx.x * blockDim.x + threadIdx.x;
    if (i < n) dst[i] = f2bf(src[i]);
}

__global__ void k_conv_h2o(const float* __restrict__ h2o_w,
                           unsigned short* __restrict__ dst) {
    int i = blockIdx.x * blockDim.x + threadIdx.x;
    if (i >= C_ * H_) return;
    int o = i / H_, k = i % H_;
    dst[i] = f2bf(h2o_w[(size_t)o * (H_ + L_) + k]);
}

__global__ void k_swish_embed(const float* __restrict__ embed,
                              float* __restrict__ se) {
    int i = blockIdx.x * blockDim.x + threadIdx.x;
    if (i >= C_ * H_) return;
    float e = embed[i];
    se[i] = e * sigm(e);
}

// Etab[ch, j] = sum_k se[ch,k] * w_ih0[j, k]
__global__ void k_etab(const float* __restrict__ se,
                       const float* __restrict__ w_ih0,
                       float* __restrict__ etab) {
    int i = blockIdx.x * blockDim.x + threadIdx.x;
    if (i >= C_ * G_) return;
    int ch = i / G_, j = i % G_;
    const float* sr = se + (size_t)ch * H_;
    const float* wr = w_ih0 + (size_t)j * (H_ + L_);
    float s = 0.f;
    for (int k = 0; k < H_; ++k) s += sr[k] * wr[k];
    etab[i] = s;
}

// giz0[b, j] = b_ih0[j] + sum_l z[b,l] * w_ih0[j, H+l]
__global__ void k_giz0(const float* __restrict__ z,
                       const float* __restrict__ w_ih0,
                       const float* __restrict__ b_ih0,
                       float* __restrict__ giz0) {
    int i = blockIdx.x * blockDim.x + threadIdx.x;
    if (i >= B_ * G_) return;
    int b = i / G_, j = i % G_;
    const float* zr = z + (size_t)b * L_;
    const float* wr = w_ih0 + (size_t)j * (H_ + L_) + H_;
    float s = b_ih0[j];
    for (int l = 0; l < L_; ++l) s += zr[l] * wr[l];
    giz0[i] = s;
}

// oz[b, o] = h2o_b[o] + sum_l z[b,l] * h2o_w[o, H+l]
__global__ void k_oz(const float* __restrict__ z,
                     const float* __restrict__ h2o_w,
                     const float* __restrict__ h2o_b,
                     float* __restrict__ oz) {
    int i = blockIdx.x * blockDim.x + threadIdx.x;
    if (i >= B_ * C_) return;
    int b = i / C_, o = i % C_;
    const float* zr = z + (size_t)b * L_;
    const float* wr = h2o_w + (size_t)o * (H_ + L_) + H_;
    float s = h2o_b[o];
    for (int l = 0; l < L_; ++l) s += zr[l] * wr[l];
    oz[i] = s;
}

// h0 = z @ z2h_w.T + z2h_b -> both layer states (fp32 + bf16)
__global__ void k_h0(const float* __restrict__ z,
                     const float* __restrict__ z2h_w,
                     const float* __restrict__ z2h_b,
                     float* __restrict__ h1f, float* __restrict__ h2f,
                     unsigned short* __restrict__ h1b, unsigned short* __restrict__ h2b) {
    int i = blockIdx.x * blockDim.x + threadIdx.x;
    if (i >= B_ * H_) return;
    int b = i >> 10, j = i & (H_ - 1);
    const float* zr = z + (size_t)b * L_;
    const float* wr = z2h_w + (size_t)j * L_;
    float s = z2h_b[j];
    for (int l = 0; l < L_; ++l) s += zr[l] * wr[l];
    h1f[i] = s; h2f[i] = s;
    unsigned short bf = f2bf(s);
    h1b[i] = bf; h2b[i] = bf;
}

__global__ void k_init_c(int* __restrict__ cvec) {
    int i = blockIdx.x * blockDim.x + threadIdx.x;
    if (i < B_) cvec[i] = 0;   // SOS
}

// ------------------------------- per-step kernels --------------------------

#define NS0 96    // cell0 tile: 3 gates * 2 k-halves * 16 rows, 32B slots (3KB)
#define NS1 192   // cell1 tile: (wih1 + whh1) * 3 gates * 2 * 16 rows   (6KB)

// GRU cell 0: gh0 = h1 @ w_hh0.T (WMMA, LDS-staged B), gi0 = Etab[c] + giz0.
// grid (H/16, B/128), block 256 (8 waves stacked along batch).
__global__ void __launch_bounds__(256)
k_gru_cell0(const unsigned short* __restrict__ h1b_in,
            const float* __restrict__ h1f_in,
            const unsigned short* __restrict__ whh0,   // bf16 [3H, H]
            const float* __restrict__ bhh0,
            const float* __restrict__ etab,            // [C, 3H]
            const float* __restrict__ giz0,            // [B, 3H]
            const int* __restrict__ cvec,
            float* __restrict__ h1f_out,
            unsigned short* __restrict__ h1b_out) {
    __shared__ unsigned short ldsB[3 * NS0 * 16];      // 3 buffers x 3KB

    const int tid   = threadIdx.x;
    const int lane  = tid & 31;
    const int wave  = tid >> 5;
    const int nn    = lane & 15;
    const int hi    = lane >> 4;
    const int jt    = blockIdx.x * 16;
    const int mtile = (blockIdx.y * 8 + wave) * 16;
    const int arow  = mtile + nn;

    // async copy of one 32-wide K-slice tile into buffer buf (192 x 16B chunks)
    auto issueB = [&](int buf, int k0) {
        if (tid < 2 * NS0) {
            int slot = tid >> 1, half = tid & 1;
            int g = slot >> 5, kh = (slot >> 4) & 1, row = slot & 15;
            const unsigned short* src =
                whh0 + (size_t)(g * H_ + jt + row) * H_ + k0 + kh * 16 + half * 8;
            async_copy16(lds_addr_of(ldsB + (buf * NS0 + slot) * 16 + half * 8), src);
        }
    };

    issueB(0, 0);
    issueB(1, 32);

    v8f accR = {0,0,0,0,0,0,0,0};
    v8f accZ = {0,0,0,0,0,0,0,0};
    v8f accN = {0,0,0,0,0,0,0,0};

    int bc = 0;   // buffer being computed this iter
    int bp = 2;   // buffer targeted by the in-loop prefetch
    for (int it = 0; it < 32; ++it) {
        const int k0 = it * 32;
        // A fragment first: global latency overlaps the async-wait + barrier
        v16bf a = load_a_frag(h1b_in, arow, H_, k0, lane);
        wait_async_le1();          // own tile `it` copies complete
        __syncthreads();           // everyone's tile `it` visible
        const unsigned short* bb = ldsB + bc * NS0 * 16;
        v16bf br = lds_b_frag(bb, (0 * 2 + hi) * 16 + nn);
        v16bf bz = lds_b_frag(bb, (1 * 2 + hi) * 16 + nn);
        v16bf bn = lds_b_frag(bb, (2 * 2 + hi) * 16 + nn);
        accR = WMMA_BF16(a, br, accR);
        accZ = WMMA_BF16(a, bz, accZ);
        accN = WMMA_BF16(a, bn, accN);
        issueB(bp, ((it + 2) & 31) * 32);   // prefetch tile it+2
        bc = (bc == 2) ? 0 : bc + 1;
        bp = (bp == 2) ? 0 : bp + 1;
    }

    const int   col = jt + nn;
    const float bR = bhh0[col], bZ = bhh0[H_ + col], bN = bhh0[2 * H_ + col];

#pragma unroll
    for (int v = 0; v < 8; ++v) {
        int m = mtile + v + 8 * hi;
        int ch = cvec[m];
        const float* Er = etab + (size_t)ch * G_;
        const float* gz = giz0 + (size_t)m * G_;
        float i_r = Er[col]          + gz[col];
        float i_z = Er[H_ + col]     + gz[H_ + col];
        float i_n = Er[2 * H_ + col] + gz[2 * H_ + col];
        float r  = sigm(i_r + (accR[v] + bR));
        float zg = sigm(i_z + (accZ[v] + bZ));
        float nl = tanhf(i_n + r * (accN[v] + bN));
        float hp = h1f_in[(size_t)m * H_ + col];
        float hn = (1.0f - zg) * nl + zg * hp;
        h1f_out[(size_t)m * H_ + col] = hn;
        h1b_out[(size_t)m * H_ + col] = f2bf(hn);
    }
}

// GRU cell 1: gi1 = h1' @ w_ih1.T and gh1 = h2 @ w_hh1.T, fused gates.
__global__ void __launch_bounds__(256)
k_gru_cell1(const unsigned short* __restrict__ h1b_new,
            const unsigned short* __restrict__ h2b_in,
            const float* __restrict__ h2f_in,
            const unsigned short* __restrict__ wih1,   // bf16 [3H, H]
            const unsigned short* __restrict__ whh1,   // bf16 [3H, H]
            const float* __restrict__ bih1,
            const float* __restrict__ bhh1,
            float* __restrict__ h2f_out,
            unsigned short* __restrict__ h2b_out) {
    __shared__ unsigned short ldsB[3 * NS1 * 16];      // 3 buffers x 6KB

    const int tid   = threadIdx.x;
    const int lane  = tid & 31;
    const int nn    = lane & 15;
    const int hi    = lane >> 4;
    const int wave  = tid >> 5;
    const int jt    = blockIdx.x * 16;
    const int mtile = (blockIdx.y * 8 + wave) * 16;
    const int arow  = mtile + nn;

    // threads 0..191 each copy one full 32B slot (2 async b128 ops) -> every
    // issuing wave posts exactly 2 ops per tile; uniform s_wait_asynccnt 2.
    auto issueB = [&](int buf, int k0) {
        if (tid < NS1) {
            int slot = tid;
            const unsigned short* W = (slot < NS0) ? wih1 : whh1;
            int s = (slot < NS0) ? slot : slot - NS0;
            int g = s >> 5, kh = (s >> 4) & 1, row = s & 15;
            const unsigned short* src =
                W + (size_t)(g * H_ + jt + row) * H_ + k0 + kh * 16;
            unsigned int dst = lds_addr_of(ldsB + (buf * NS1 + slot) * 16);
            async_copy16(dst, src);
            async_copy16(dst + 16, src + 8);
        }
    };

    issueB(0, 0);
    issueB(1, 32);

    v8f aiR = {0,0,0,0,0,0,0,0}, aiZ = {0,0,0,0,0,0,0,0}, aiN = {0,0,0,0,0,0,0,0};
    v8f ahR = {0,0,0,0,0,0,0,0}, ahZ = {0,0,0,0,0,0,0,0}, ahN = {0,0,0,0,0,0,0,0};

    int bc = 0;
    int bp = 2;
    for (int it = 0; it < 32; ++it) {
        const int k0 = it * 32;
        v16bf a1 = load_a_frag(h1b_new, arow, H_, k0, lane);
        v16bf a2 = load_a_frag(h2b_in,  arow, H_, k0, lane);
        wait_async_le2();          // own tile `it` (2 ops) complete
        __syncthreads();
        const unsigned short* bb = ldsB + bc * NS1 * 16;
        v16bf b;
        b = lds_b_frag(bb, (0 * 2 + hi) * 16 + nn);        aiR = WMMA_BF16(a1, b, aiR);
        b = lds_b_frag(bb, (1 * 2 + hi) * 16 + nn);        aiZ = WMMA_BF16(a1, b, aiZ);
        b = lds_b_frag(bb, (2 * 2 + hi) * 16 + nn);        aiN = WMMA_BF16(a1, b, aiN);
        b = lds_b_frag(bb, NS0 + (0 * 2 + hi) * 16 + nn);  ahR = WMMA_BF16(a2, b, ahR);
        b = lds_b_frag(bb, NS0 + (1 * 2 + hi) * 16 + nn);  ahZ = WMMA_BF16(a2, b, ahZ);
        b = lds_b_frag(bb, NS0 + (2 * 2 + hi) * 16 + nn);  ahN = WMMA_BF16(a2, b, ahN);
        issueB(bp, ((it + 2) & 31) * 32);
        bc = (bc == 2) ? 0 : bc + 1;
        bp = (bp == 2) ? 0 : bp + 1;
    }

    const int   col = jt + nn;
    const float biR = bih1[col], biZ = bih1[H_ + col], biN = bih1[2 * H_ + col];
    const float bhR = bhh1[col], bhZ = bhh1[H_ + col], bhN = bhh1[2 * H_ + col];

#pragma unroll
    for (int v = 0; v < 8; ++v) {
        int m = mtile + v + 8 * hi;
        float r  = sigm((aiR[v] + biR) + (ahR[v] + bhR));
        float zg = sigm((aiZ[v] + biZ) + (ahZ[v] + bhZ));
        float nl = tanhf((aiN[v] + biN) + r * (ahN[v] + bhN));
        float hp = h2f_in[(size_t)m * H_ + col];
        float hn = (1.0f - zg) * nl + zg * hp;
        h2f_out[(size_t)m * H_ + col] = hn;
        h2b_out[(size_t)m * H_ + col] = f2bf(hn);
    }
}

// logits = h2' @ h2o[:, :H].T + oz ; store d_out[:, t, :]; argmax -> cvec
__global__ void __launch_bounds__(256)
k_out_argmax(const unsigned short* __restrict__ h2b_new,
             const unsigned short* __restrict__ h2ob,  // bf16 [C, H]
             const float* __restrict__ oz,             // [B, C]
             float* __restrict__ out,                  // [B, T, C]
             int t,
             int* __restrict__ cvec) {
    __shared__ float lds[16 * C_];

    const int lane  = threadIdx.x & 31;
    const int wave  = threadIdx.x >> 5;
    const int nn    = lane & 15;
    const int hi    = lane >> 4;
    const int mtile = blockIdx.x * 16;
    const int nt    = wave * 16;
    const int arow  = mtile + nn;

    v8f acc = {0,0,0,0,0,0,0,0};
    for (int k0 = 0; k0 < H_; k0 += 32) {
        v16bf a = load_a_frag(h2b_new, arow, H_, k0, lane);
        v16bf b = load_b_frag(h2ob, nt + nn, H_, k0, lane);
        acc = WMMA_BF16(a, b, acc);
    }

    const int col = nt + nn;
#pragma unroll
    for (int v = 0; v < 8; ++v) {
        int ml = v + 8 * hi;
        int m  = mtile + ml;
        float val = acc[v] + oz[(size_t)m * C_ + col];
        lds[ml * C_ + col] = val;
        out[((size_t)m * T_ + t) * C_ + col] = val;
    }
    __syncthreads();

    if (threadIdx.x < 16) {
        int r = threadIdx.x;
        const float* row = lds + r * C_;
        float best = row[0];
        int   bi   = 0;
        for (int o = 1; o < C_; ++o) {
            float v = row[o];
            if (v > best) { best = v; bi = o; }   // first-max tie-break
        }
        cvec[mtile + r] = bi;
    }
}

// ------------------------------- host side ---------------------------------

extern "C" void kernel_launch(void* const* d_in, const int* in_sizes, int n_in,
                              void* d_out, int out_size, void* d_ws, size_t ws_size,
                              hipStream_t stream) {
    (void)in_sizes; (void)n_in; (void)out_size; (void)ws_size;

    const float* z      = (const float*)d_in[0];
    const float* embed  = (const float*)d_in[1];
    const float* z2h_w  = (const float*)d_in[2];
    const float* z2h_b  = (const float*)d_in[3];
    const float* w_ih0  = (const float*)d_in[4];
    const float* w_hh0  = (const float*)d_in[5];
    const float* b_ih0  = (const float*)d_in[6];
    const float* b_hh0  = (const float*)d_in[7];
    const float* w_ih1  = (const float*)d_in[8];
    const float* w_hh1  = (const float*)d_in[9];
    const float* b_ih1  = (const float*)d_in[10];
    const float* b_hh1  = (const float*)d_in[11];
    const float* h2o_w  = (const float*)d_in[12];
    const float* h2o_b  = (const float*)d_in[13];
    float* out = (float*)d_out;

    char* w = (char*)d_ws;
    auto alloc = [&](size_t bytes) -> void* {
        void* p = (void*)w;
        w += (bytes + 255) & ~(size_t)255;
        return p;
    };
    float* h1f[2];  float* h2f[2];
    unsigned short* h1b[2]; unsigned short* h2b[2];
    h1f[0] = (float*)alloc((size_t)B_ * H_ * 4);
    h1f[1] = (float*)alloc((size_t)B_ * H_ * 4);
    h2f[0] = (float*)alloc((size_t)B_ * H_ * 4);
    h2f[1] = (float*)alloc((size_t)B_ * H_ * 4);
    h1b[0] = (unsigned short*)alloc((size_t)B_ * H_ * 2);
    h1b[1] = (unsigned short*)alloc((size_t)B_ * H_ * 2);
    h2b[0] = (unsigned short*)alloc((size_t)B_ * H_ * 2);
    h2b[1] = (unsigned short*)alloc((size_t)B_ * H_ * 2);
    float* giz0 = (float*)alloc((size_t)B_ * G_ * 4);
    float* etab = (float*)alloc((size_t)C_ * G_ * 4);
    float* se   = (float*)alloc((size_t)C_ * H_ * 4);
    float* oz   = (float*)alloc((size_t)B_ * C_ * 4);
    unsigned short* whh0b = (unsigned short*)alloc((size_t)G_ * H_ * 2);
    unsigned short* wih1b = (unsigned short*)alloc((size_t)G_ * H_ * 2);
    unsigned short* whh1b = (unsigned short*)alloc((size_t)G_ * H_ * 2);
    unsigned short* h2ob  = (unsigned short*)alloc((size_t)C_ * H_ * 2);
    int* cvec = (int*)alloc((size_t)B_ * 4);

    // ---- one-time setup (re-run every call: deterministic, stateless) ----
    const int TB = 256;
    k_convert_bf16<<<(G_ * H_ + TB - 1) / TB, TB, 0, stream>>>(w_hh0, whh0b, G_ * H_);
    k_convert_bf16<<<(G_ * H_ + TB - 1) / TB, TB, 0, stream>>>(w_ih1, wih1b, G_ * H_);
    k_convert_bf16<<<(G_ * H_ + TB - 1) / TB, TB, 0, stream>>>(w_hh1, whh1b, G_ * H_);
    k_conv_h2o<<<(C_ * H_ + TB - 1) / TB, TB, 0, stream>>>(h2o_w, h2ob);
    k_swish_embed<<<(C_ * H_ + TB - 1) / TB, TB, 0, stream>>>(embed, se);
    k_etab<<<(C_ * G_ + TB - 1) / TB, TB, 0, stream>>>(se, w_ih0, etab);
    k_giz0<<<(B_ * G_ + TB - 1) / TB, TB, 0, stream>>>(z, w_ih0, b_ih0, giz0);
    k_oz<<<(B_ * C_ + TB - 1) / TB, TB, 0, stream>>>(z, h2o_w, h2o_b, oz);
    k_h0<<<(B_ * H_ + TB - 1) / TB, TB, 0, stream>>>(z, z2h_w, z2h_b,
                                                     h1f[0], h2f[0], h1b[0], h2b[0]);
    k_init_c<<<(B_ + TB - 1) / TB, TB, 0, stream>>>(cvec);

    // ---- 64 decode steps ----
    dim3 grid_gru(H_ / 16, B_ / (16 * 8));   // (64, 32)
    for (int t = 0; t < T_; ++t) {
        int pi = t & 1;
        int po = 1 - pi;
        k_gru_cell0<<<grid_gru, 256, 0, stream>>>(
            h1b[pi], h1f[pi], whh0b, b_hh0, etab, giz0, cvec, h1f[po], h1b[po]);
        k_gru_cell1<<<grid_gru, 256, 0, stream>>>(
            h1b[po], h2b[pi], h2f[pi], wih1b, whh1b, b_ih1, b_hh1, h2f[po], h2b[po]);
        k_out_argmax<<<B_ / 16, 256, 0, stream>>>(
            h2b[po], h2ob, oz, out, t, cvec);
    }
}
